// TextGRUClassifier_74062416053148
// MI455X (gfx1250) — compile-verified, matrix operand order checked by reading, not run on hardware
//
#include <hip/hip_runtime.h>
#include <hip/hip_bf16.h>
#include <stdint.h>

// Problem constants (from reference)
#define TSEQ  512
#define BATCH 256
#define EDIM  128
#define HDIM  160
#define G3H   480   // 3*H

typedef __attribute__((ext_vector_type(16))) __bf16 v16bf;
typedef __attribute__((ext_vector_type(8)))  float  v8f;

union Frag {
  v16bf v;
  int4  q[2];
};

__device__ __forceinline__ unsigned short f2bf(float f) {
  unsigned int u = __float_as_uint(f);
  u += 0x7FFFu + ((u >> 16) & 1u);   // round-to-nearest-even
  return (unsigned short)(u >> 16);
}

__device__ __forceinline__ float sigmoid_fast(float x) {
  return 1.0f / (1.0f + __expf(-x));
}
__device__ __forceinline__ float tanh_fast(float x) {
  float e = __expf(-2.0f * x);
  return (1.0f - e) / (1.0f + e);
}

// ---------------------------------------------------------------------------
// fp32 -> bf16 weight conversion
// ---------------------------------------------------------------------------
__global__ void cast_f32_bf16_kernel(const float* __restrict__ in,
                                     unsigned short* __restrict__ out, int n) {
  int i = blockIdx.x * blockDim.x + threadIdx.x;
  if (i < n) out[i] = f2bf(in[i]);
}

// ---------------------------------------------------------------------------
// Embedding gather + cast: xs[t,b,e] = bf16(emb[x[b,t], e])
// grid.x = T*B (t = blk>>8, b = blk&255), block = 128 threads (one per e)
// ---------------------------------------------------------------------------
__global__ void embed_gather_kernel(const int* __restrict__ x,
                                    const float* __restrict__ emb,
                                    unsigned short* __restrict__ xs) {
  const int t = blockIdx.x >> 8;
  const int b = blockIdx.x & 255;
  const int e = threadIdx.x;
  const int idx = x[b * TSEQ + t];
  xs[(size_t)blockIdx.x * EDIM + e] = f2bf(emb[(size_t)idx * EDIM + e]);
}

// ---------------------------------------------------------------------------
// C[M,N] = A[M,K] @ W[N,K]^T + bias[N]    (A,W bf16 row-major; C fp32)
// One wave per 16x16 C tile. Block = 8 waves => 8 consecutive M tiles.
// grid = (M/128, N/16). K multiple of 32.
// A frag (16x32 bf16): lane holds row=lane&15, k = (lane>>4)*8 + {0..7, 16..23}
// B frag (32x16 bf16): lane holds col=lane&15, same k split  (W row = col)
// C frag (v8f): element r -> (m = (lane>>4)*8 + r, n = lane&15)
// ---------------------------------------------------------------------------
__global__ __launch_bounds__(256)
void gemm_bias_wmma_kernel(const unsigned short* __restrict__ A,
                           const unsigned short* __restrict__ W,
                           const float* __restrict__ bias,
                           float* __restrict__ C,
                           int M, int N, int K) {
  const int lane  = threadIdx.x & 31;
  const int wave  = threadIdx.x >> 5;
  const int lrow  = lane & 15;
  const int khalf = (lane >> 4) << 3;           // 0 or 8
  const int mtile = blockIdx.x * 8 + wave;
  const int n0    = blockIdx.y * 16;
  const int m0    = mtile * 16;
  const int ncol  = n0 + lrow;

  v8f acc;
  {
    const float bv = bias[ncol];
#pragma unroll
    for (int r = 0; r < 8; ++r) acc[r] = bv;
  }

  const unsigned short* arow = A + (size_t)(m0 + lrow) * K;
  const unsigned short* wrow = W + (size_t)ncol * K;
  for (int kk = 0; kk < K; kk += 32) {
    Frag a, b;
    a.q[0] = *(const int4*)(arow + kk + khalf);
    a.q[1] = *(const int4*)(arow + kk + khalf + 16);
    b.q[0] = *(const int4*)(wrow + kk + khalf);
    b.q[1] = *(const int4*)(wrow + kk + khalf + 16);
    acc = __builtin_amdgcn_wmma_f32_16x16x32_bf16(false, a.v, false, b.v,
                                                  (short)0, acc, false, false);
  }

#pragma unroll
  for (int r = 0; r < 8; ++r)
    C[(size_t)(m0 + khalf + r) * N + ncol] = acc[r];
}

// ---------------------------------------------------------------------------
// One GRU direction scan over T for a 16-row batch tile.
// grid = (16 batch tiles, 2 directions), block = 320 threads (10 waves).
// Whh (480x160 bf16) lives in VGPRs: 15 fragments/wave for the whole scan.
// Per step: gh = h @ Whh^T + bhh via 15 WMMA/wave, stage in LDS, gate math,
// new h written back to LDS (fp32 + bf16 copies).
// ---------------------------------------------------------------------------
__global__ __launch_bounds__(320)
void gru_scan_kernel(const float* __restrict__ gi_f,   // [T,B,480] fp32 (incl. bih)
                     const float* __restrict__ gi_b,
                     const unsigned short* __restrict__ whh_f,  // [480,160] bf16
                     const unsigned short* __restrict__ whh_b,
                     const float* __restrict__ bhh_f,  // [480]
                     const float* __restrict__ bhh_b,
                     unsigned short* __restrict__ y,   // [T,B,320] bf16 (layer0 out)
                     float* __restrict__ hfinal,       // [B,320] fp32 or nullptr
                     int store_y) {
  __shared__ float          gh[16][G3H];     // 30720 B
  __shared__ float          hbuf[16][HDIM];  // 10240 B
  __shared__ unsigned short hbf[16][HDIM];   //  5120 B

  const int dir   = blockIdx.y;
  const int b0    = blockIdx.x * 16;
  const int lane  = threadIdx.x & 31;
  const int wave  = threadIdx.x >> 5;          // 0..9
  const int lrow  = lane & 15;
  const int khalf = (lane >> 4) << 3;          // 0 or 8 (K half / M half)

  const float* gi            = dir ? gi_b  : gi_f;
  const unsigned short* whh  = dir ? whh_b : whh_f;
  const float* bhh           = dir ? bhh_b : bhh_f;

  // Preload recurrent weights into registers: wave owns N tiles wave*3+{0,1,2}
  Frag  bw[3][5];
  float bb[3];
#pragma unroll
  for (int j = 0; j < 3; ++j) {
    const int ncol = (wave * 3 + j) * 16 + lrow;
    bb[j] = bhh[ncol];
    const unsigned short* wr = whh + (size_t)ncol * HDIM;
#pragma unroll
    for (int k = 0; k < 5; ++k) {
      bw[j][k].q[0] = *(const int4*)(wr + k * 32 + khalf);
      bw[j][k].q[1] = *(const int4*)(wr + k * 32 + khalf + 16);
    }
  }

  // Elementwise-phase mapping: 320 threads x 8 elements = 16 x 160
  const int m     = threadIdx.x / 20;
  const int jb    = (threadIdx.x % 20) * 8;
  const int bglob = b0 + m;

  // h0 = 0
#pragma unroll
  for (int e = 0; e < 8; ++e) { hbuf[m][jb + e] = 0.0f; hbf[m][jb + e] = 0; }
  __syncthreads();

  for (int t = 0; t < TSEQ; ++t) {
    const int tt = dir ? (TSEQ - 1 - t) : t;

    // ---- gh = h @ Whh^T + bhh ----
    v8f acc[3];
#pragma unroll
    for (int j = 0; j < 3; ++j)
#pragma unroll
      for (int r = 0; r < 8; ++r) acc[j][r] = bb[j];

#pragma unroll
    for (int k = 0; k < 5; ++k) {
      Frag a;
      const unsigned short* hr = &hbf[lrow][k * 32 + khalf];
      a.q[0] = *(const int4*)(hr);
      a.q[1] = *(const int4*)(hr + 16);
#pragma unroll
      for (int j = 0; j < 3; ++j)
        acc[j] = __builtin_amdgcn_wmma_f32_16x16x32_bf16(
            false, a.v, false, bw[j][k].v, (short)0, acc[j], false, false);
    }
#pragma unroll
    for (int j = 0; j < 3; ++j) {
      const int ncol = (wave * 3 + j) * 16 + lrow;
#pragma unroll
      for (int r = 0; r < 8; ++r)
        gh[khalf + r][ncol] = acc[j][r];
    }
    __syncthreads();

    // ---- gates + state update ----
    const float* gptr = gi + ((size_t)tt * BATCH + bglob) * G3H + jb;
    alignas(16) unsigned short packed[8];
#pragma unroll
    for (int e = 0; e < 8; ++e) {
      const float ir  = gptr[e];
      const float iz  = gptr[HDIM + e];
      const float inn = gptr[2 * HDIM + e];
      const float r_  = sigmoid_fast(ir + gh[m][jb + e]);
      const float z_  = sigmoid_fast(iz + gh[m][HDIM + jb + e]);
      const float n_  = tanh_fast(inn + r_ * gh[m][2 * HDIM + jb + e]);
      const float hp  = hbuf[m][jb + e];
      const float hn  = (1.0f - z_) * n_ + z_ * hp;
      hbuf[m][jb + e] = hn;
      const unsigned short hb = f2bf(hn);
      hbf[m][jb + e] = hb;
      packed[e] = hb;
    }
    if (store_y) {
      *(int4*)(y + ((size_t)tt * BATCH + bglob) * (2 * HDIM) + dir * HDIM + jb) =
          *(const int4*)packed;
    }
    __syncthreads();
  }

  if (hfinal) {
#pragma unroll
    for (int e = 0; e < 8; ++e)
      hfinal[(size_t)bglob * (2 * HDIM) + dir * HDIM + jb + e] = hbuf[m][jb + e];
  }
}

// ---------------------------------------------------------------------------
// Head: out[b] = fc2 @ relu(fc1 @ h[b] + b1) + b2.  grid=256, block=128.
// ---------------------------------------------------------------------------
__global__ __launch_bounds__(128)
void fc_head_kernel(const float* __restrict__ hfin,
                    const float* __restrict__ w1, const float* __restrict__ b1,
                    const float* __restrict__ w2, const float* __restrict__ b2,
                    float* __restrict__ out) {
  __shared__ float sh[128];
  const int b = blockIdx.x;
  const int j = threadIdx.x;
  const float* hrow = hfin + (size_t)b * (2 * HDIM);
  const float* wrow = w1 + (size_t)j * (2 * HDIM);
  float s = b1[j];
  for (int k = 0; k < 2 * HDIM; k += 4) {
    float4 hv = *(const float4*)(hrow + k);
    float4 wv = *(const float4*)(wrow + k);
    s = fmaf(hv.x, wv.x, s); s = fmaf(hv.y, wv.y, s);
    s = fmaf(hv.z, wv.z, s); s = fmaf(hv.w, wv.w, s);
  }
  s = fmaxf(s, 0.0f);
  sh[j] = s * w2[j];
  __syncthreads();
  for (int off = 64; off > 0; off >>= 1) {
    if (j < off) sh[j] += sh[j + off];
    __syncthreads();
  }
  if (j == 0) out[b] = sh[0] + b2[0];
}

// ---------------------------------------------------------------------------
extern "C" void kernel_launch(void* const* d_in, const int* in_sizes, int n_in,
                              void* d_out, int out_size, void* d_ws, size_t ws_size,
                              hipStream_t stream) {
  const int*   x     = (const int*)d_in[0];
  const float* emb   = (const float*)d_in[1];
  const float* Wih0f = (const float*)d_in[2];
  const float* Whh0f = (const float*)d_in[3];
  const float* bih0f = (const float*)d_in[4];
  const float* bhh0f = (const float*)d_in[5];
  const float* Wih0b = (const float*)d_in[6];
  const float* Whh0b = (const float*)d_in[7];
  const float* bih0b = (const float*)d_in[8];
  const float* bhh0b = (const float*)d_in[9];
  const float* Wih1f = (const float*)d_in[10];
  const float* Whh1f = (const float*)d_in[11];
  const float* bih1f = (const float*)d_in[12];
  const float* bhh1f = (const float*)d_in[13];
  const float* Wih1b = (const float*)d_in[14];
  const float* Whh1b = (const float*)d_in[15];
  const float* bih1b = (const float*)d_in[16];
  const float* bhh1b = (const float*)d_in[17];
  const float* fc1w  = (const float*)d_in[18];
  const float* fc1b  = (const float*)d_in[19];
  const float* fc2w  = (const float*)d_in[20];
  const float* fc2b  = (const float*)d_in[21];
  float* out = (float*)d_out;

  // Workspace carve (256B aligned)
  char* ws = (char*)d_ws;
  size_t off = 0;
  auto carve = [&](size_t bytes) -> char* {
    char* p = ws + off;
    off += (bytes + 255) & ~(size_t)255;
    return p;
  };
  unsigned short* wih0f_h = (unsigned short*)carve((size_t)G3H * EDIM * 2);
  unsigned short* wih0b_h = (unsigned short*)carve((size_t)G3H * EDIM * 2);
  unsigned short* whh0f_h = (unsigned short*)carve((size_t)G3H * HDIM * 2);
  unsigned short* whh0b_h = (unsigned short*)carve((size_t)G3H * HDIM * 2);
  unsigned short* whh1f_h = (unsigned short*)carve((size_t)G3H * HDIM * 2);
  unsigned short* whh1b_h = (unsigned short*)carve((size_t)G3H * HDIM * 2);
  unsigned short* wih1f_h = (unsigned short*)carve((size_t)G3H * 2 * HDIM * 2);
  unsigned short* wih1b_h = (unsigned short*)carve((size_t)G3H * 2 * HDIM * 2);
  unsigned short* xs      = (unsigned short*)carve((size_t)TSEQ * BATCH * EDIM * 2);
  unsigned short* y0      = (unsigned short*)carve((size_t)TSEQ * BATCH * 2 * HDIM * 2);
  float* gi_f   = (float*)carve((size_t)TSEQ * BATCH * G3H * 4);
  float* gi_b   = (float*)carve((size_t)TSEQ * BATCH * G3H * 4);
  float* hfinal = (float*)carve((size_t)BATCH * 2 * HDIM * 4);

  // 1. bf16 weight conversion
  auto cast = [&](const float* src, unsigned short* dst, int n) {
    cast_f32_bf16_kernel<<<(n + 255) / 256, 256, 0, stream>>>(src, dst, n);
  };
  cast(Wih0f, wih0f_h, G3H * EDIM);
  cast(Wih0b, wih0b_h, G3H * EDIM);
  cast(Whh0f, whh0f_h, G3H * HDIM);
  cast(Whh0b, whh0b_h, G3H * HDIM);
  cast(Whh1f, whh1f_h, G3H * HDIM);
  cast(Whh1b, whh1b_h, G3H * HDIM);
  cast(Wih1f, wih1f_h, G3H * 2 * HDIM);
  cast(Wih1b, wih1b_h, G3H * 2 * HDIM);

  // 2. embedding gather -> xs [T,B,E] bf16
  embed_gather_kernel<<<TSEQ * BATCH, EDIM, 0, stream>>>(x, emb, xs);

  const int Mrows  = TSEQ * BATCH;          // 131072
  const dim3 ggrid(Mrows / 128, G3H / 16);  // (1024, 30)

  // 3. layer-0 input gates: gi = xs @ Wih^T + bih   (K = 128)
  gemm_bias_wmma_kernel<<<ggrid, 256, 0, stream>>>(xs, wih0f_h, bih0f, gi_f,
                                                   Mrows, G3H, EDIM);
  gemm_bias_wmma_kernel<<<ggrid, 256, 0, stream>>>(xs, wih0b_h, bih0b, gi_b,
                                                   Mrows, G3H, EDIM);

  // 4. layer-0 bidirectional scan -> y0 [T,B,320] bf16
  gru_scan_kernel<<<dim3(BATCH / 16, 2), 320, 0, stream>>>(
      gi_f, gi_b, whh0f_h, whh0b_h, bhh0f, bhh0b, y0, nullptr, 1);

  // 5. layer-1 input gates: gi = y0 @ Wih1^T + bih1   (K = 320)
  gemm_bias_wmma_kernel<<<ggrid, 256, 0, stream>>>(y0, wih1f_h, bih1f, gi_f,
                                                   Mrows, G3H, 2 * HDIM);
  gemm_bias_wmma_kernel<<<ggrid, 256, 0, stream>>>(y0, wih1b_h, bih1b, gi_b,
                                                   Mrows, G3H, 2 * HDIM);

  // 6. layer-1 scan, keep only final hidden states -> hfinal [B,320]
  gru_scan_kernel<<<dim3(BATCH / 16, 2), 320, 0, stream>>>(
      gi_f, gi_b, whh1f_h, whh1b_h, bhh1f, bhh1b, y0, hfinal, 0);

  // 7. MLP head -> out [B,1]
  fc_head_kernel<<<BATCH, 128, 0, stream>>>(hfinal, fc1w, fc1b, fc2w, fc2b, out);
}
